// CustomGraphNet_26439818674255
// MI455X (gfx1250) — compile-verified, matrix operand order checked by reading, not run on hardware
//
#include <hip/hip_runtime.h>
#include <hip/hip_bf16.h>
#include <stdint.h>

// ---------------------------------------------------------------------------
// Problem constants (match reference)
// ---------------------------------------------------------------------------
#define N_NODES_C 50000
#define N_EDGES_C 400000
#define LATENT_C  128
#define LN_EPS_C  1e-5f

typedef __attribute__((ext_vector_type(16))) _Float16 v16h;
typedef __attribute__((ext_vector_type(8)))  float    v8f;

union FragU { v16h h; unsigned u[8]; };

__device__ inline v8f vzero8() {
  v8f z;
#pragma unroll
  for (int i = 0; i < 8; ++i) z[i] = 0.f;
  return z;
}

__device__ inline v8f wmma_f16(const FragU& a, const FragU& b, v8f c) {
  return __builtin_amdgcn_wmma_f32_16x16x32_f16(
      /*neg_a=*/false, a.h, /*neg_b=*/false, b.h,
      /*c_mod=*/(short)0, c, /*reuse_a=*/false, /*reuse_b=*/false);
}

// A fragment (16x32 f16) for rows of an LDS tile (row-major halfs, stride AS).
// Layout per ISA 7.12.2: lane = m + 16*half; VGPR v holds packed pair at
// k = kt*32 + ((v&4)?16:0) + half*8 + (v&3)*2.
__device__ inline FragU load_a_lds(const _Float16* aT, int AS, int lane, int kt) {
  FragU f;
  const int m = lane & 15, half = lane >> 4;
  const _Float16* base = aT + m * AS + kt * 32 + half * 8;
  const unsigned* p0 = (const unsigned*)(base);
  const unsigned* p1 = (const unsigned*)(base + 16);
#pragma unroll
  for (int v = 0; v < 4; ++v) f.u[v] = p0[v];
#pragma unroll
  for (int v = 0; v < 4; ++v) f.u[4 + v] = p1[v];
  return f;
}

// B fragment from pre-packed weights: dword index ((kt*NT+nt)*32+lane)*8+v.
__device__ inline FragU load_b_packed(const unsigned* pw, int NT, int kt, int nt, int lane) {
  FragU f;
  const unsigned* p = pw + (((size_t)(kt * NT + nt) * 32 + lane) << 3);
#pragma unroll
  for (int v = 0; v < 8; ++v) f.u[v] = p[v];
  return f;
}

// Order-preserving float<->int transform for atomic max on floats.
__device__ inline int f2ord(float f) {
  int i = __float_as_int(f);
  return (i >= 0) ? i : (i ^ 0x7fffffff);
}
__device__ inline float ord2f(int i) {
  return __int_as_float((i >= 0) ? i : (i ^ 0x7fffffff));
}

// ---------------------------------------------------------------------------
// Weight packing: f32 row-major W[K][Nout] -> WMMA B-fragment order (f16)
// ---------------------------------------------------------------------------
__global__ __launch_bounds__(256) void pack_weights_kernel(
    const float* __restrict__ W, unsigned* __restrict__ dst,
    int K, int Nout, int NT, int total) {
  int id = blockIdx.x * 256 + threadIdx.x;
  if (id >= total) return;
  int v    = id & 7;
  int lane = (id >> 3) & 31;
  int tile = id >> 8;
  int nt = tile % NT;
  int kt = tile / NT;
  int n    = (lane & 15) + nt * 16;
  int half = lane >> 4;
  int k = kt * 32 + ((v & 4) ? 16 : 0) + half * 8 + (v & 3) * 2;
  float f0 = (k < K && n < Nout) ? W[(size_t)k * Nout + n] : 0.f;
  float f1 = (k + 1 < K && n < Nout) ? W[(size_t)(k + 1) * Nout + n] : 0.f;
  union { _Float16 h[2]; unsigned u; } pk;
  pk.h[0] = (_Float16)f0;
  pk.h[1] = (_Float16)f1;
  dst[id] = pk.u;
}

// ---------------------------------------------------------------------------
// Fused fnet kernel: y = fnet(x) with fnet = Lin+ReLU -> ResBlock -> Lin (+LN)
// 16 rows per block, 8 waves (each wave = one 16-col output stripe).
// ---------------------------------------------------------------------------
enum { MODE_RAW = 0, MODE_LATENT = 1, MODE_CEDGE = 2, MODE_CNODE = 3 };

__global__ __launch_bounds__(256) void fnet_kernel(
    int mode, int Kin, int KtIn,
    const float* __restrict__ rawIn,
    const _Float16* __restrict__ nodeF16,
    const _Float16* __restrict__ edgeF16,
    const int* __restrict__ srcIdx,
    const int* __restrict__ dstIdx,
    const float* __restrict__ aggF32,
    const unsigned* __restrict__ pWin,  const float* __restrict__ binB,
    const unsigned* __restrict__ pW1,   const float* __restrict__ b1B,
    const unsigned* __restrict__ pW2,   const float* __restrict__ b2B,
    const unsigned* __restrict__ pWout, const float* __restrict__ boutB,
    int NTout, int Nout,
    const float* __restrict__ lnG, const float* __restrict__ lnB,
    const _Float16* __restrict__ resPtr,
    _Float16* __restrict__ outF16, float* __restrict__ outF32) {
  __shared__ __align__(16) _Float16 aT[16 * 392];
  __shared__ __align__(16) _Float16 hT[16 * 136];
  __shared__ __align__(16) _Float16 tT[16 * 136];
  __shared__ __align__(16) float    yT[16 * 132];
  __shared__ float lnS[32];

  const int tid  = threadIdx.x;
  const int wave = tid >> 5;
  const int lane = tid & 31;
  const int m0   = blockIdx.x << 4;

  int AS;
  // ---- cooperative input-tile load ----
  if (mode == MODE_RAW) {
    AS = 40;
    for (int i = tid; i < 512; i += 256) {
      int r = i >> 5, k = i & 31;
      float v = (k < Kin) ? rawIn[(size_t)(m0 + r) * Kin + k] : 0.f;
      aT[r * 40 + k] = (_Float16)v;
    }
  } else if (mode == MODE_LATENT) {
    AS = 136;
    int r = tid >> 4, ch = tid & 15;
    *(uint4*)&aT[r * 136 + ch * 8] =
        ((const uint4*)(nodeF16 + (size_t)(m0 + r) * 128))[ch];
  } else if (mode == MODE_CEDGE) {
    AS = 392;
#pragma unroll
    for (int i = 0; i < 3; ++i) {
      int u = tid + i * 256;
      int r = u / 48, pos = u % 48;
      int seg = pos >> 4, ch = pos & 15;
      int e = m0 + r;
      const _Float16* srcRow;
      if (seg == 0)      srcRow = nodeF16 + (size_t)dstIdx[e] * 128;
      else if (seg == 1) srcRow = nodeF16 + (size_t)srcIdx[e] * 128;
      else               srcRow = edgeF16 + (size_t)e * 128;
      *(uint4*)&aT[r * 392 + seg * 128 + ch * 8] = ((const uint4*)srcRow)[ch];
    }
  } else {  // MODE_CNODE: [node | agg(f32)]
    AS = 264;
#pragma unroll
    for (int i = 0; i < 2; ++i) {
      int u = tid + i * 256;
      int r = u >> 5, pos = u & 31;
      int seg = pos >> 4, ch = pos & 15;
      int gm = m0 + r;
      if (seg == 0) {
        *(uint4*)&aT[r * 264 + ch * 8] =
            ((const uint4*)(nodeF16 + (size_t)gm * 128))[ch];
      } else {
        const float* s = aggF32 + (size_t)gm * 128 + ch * 8;
        _Float16* d = &aT[r * 264 + 128 + ch * 8];
#pragma unroll
        for (int j = 0; j < 8; ++j) d[j] = (_Float16)s[j];
      }
    }
  }
  __syncthreads();

  const int nloc = lane & 15;
  const int half = lane >> 4;
  const int colw = wave * 16 + nloc;

  // ---- stage 1: h = relu(x @ Win + bin) ----
  v8f c = vzero8();
  for (int kt = 0; kt < KtIn; ++kt) {
    FragU a = load_a_lds(aT, AS, lane, kt);
    FragU b = load_b_packed(pWin, 8, kt, wave, lane);
    c = wmma_f16(a, b, c);
  }
  {
    float bv = binB[colw];
#pragma unroll
    for (int r = 0; r < 8; ++r)
      hT[(r + 8 * half) * 136 + colw] = (_Float16)fmaxf(c[r] + bv, 0.f);
  }
  __syncthreads();

  // ---- stage 2: t = relu(h @ W1 + b1) ----
  c = vzero8();
  for (int kt = 0; kt < 4; ++kt) {
    FragU a = load_a_lds(hT, 136, lane, kt);
    FragU b = load_b_packed(pW1, 8, kt, wave, lane);
    c = wmma_f16(a, b, c);
  }
  {
    float bv = b1B[colw];
#pragma unroll
    for (int r = 0; r < 8; ++r)
      tT[(r + 8 * half) * 136 + colw] = (_Float16)fmaxf(c[r] + bv, 0.f);
  }
  __syncthreads();

  // ---- stage 3: h = h + relu(t @ W2 + b2) ----
  c = vzero8();
  for (int kt = 0; kt < 4; ++kt) {
    FragU a = load_a_lds(tT, 136, lane, kt);
    FragU b = load_b_packed(pW2, 8, kt, wave, lane);
    c = wmma_f16(a, b, c);
  }
  {
    float bv = b2B[colw];
#pragma unroll
    for (int r = 0; r < 8; ++r) {
      int idx = (r + 8 * half) * 136 + colw;
      float hv = (float)hT[idx];
      hT[idx] = (_Float16)(hv + fmaxf(c[r] + bv, 0.f));
    }
  }
  __syncthreads();

  // ---- stage 4: y = h @ Wout + bout ----
  if (wave < NTout) {
    c = vzero8();
    for (int kt = 0; kt < 4; ++kt) {
      FragU a = load_a_lds(hT, 136, lane, kt);
      FragU b = load_b_packed(pWout, NTout, kt, wave, lane);
      c = wmma_f16(a, b, c);
    }
    float bv = (colw < Nout) ? boutB[colw] : 0.f;
#pragma unroll
    for (int r = 0; r < 8; ++r)
      yT[(r + 8 * half) * 132 + colw] = c[r] + bv;
  }
  __syncthreads();

  // ---- epilogue ----
  if (outF32) {  // decoder: no LN, narrow f32 output
    if (tid < 16) {
      for (int j = 0; j < Nout; ++j)
        outF32[(size_t)(m0 + tid) * Nout + j] = yT[tid * 132 + j];
    }
  } else {       // LayerNorm (+ optional residual), f16 output
    if (tid < 16) {
      float s = 0.f, s2 = 0.f;
      for (int k = 0; k < 128; ++k) {
        float v = yT[tid * 132 + k];
        s += v; s2 += v * v;
      }
      float mu  = s * (1.f / 128.f);
      float var = s2 * (1.f / 128.f) - mu * mu;
      lnS[tid]      = mu;
      lnS[16 + tid] = rsqrtf(var + LN_EPS_C);
    }
    __syncthreads();
    int r = tid >> 4, ch = tid & 15;
    float mu = lnS[r], rs = lnS[16 + r];
    union { _Float16 h[8]; uint4 q; } o;
#pragma unroll
    for (int j = 0; j < 8; ++j) {
      int col = ch * 8 + j;
      float v = (yT[r * 132 + col] - mu) * rs * lnG[col] + lnB[col];
      if (resPtr) v += (float)resPtr[(size_t)(m0 + r) * 128 + col];
      o.h[j] = (_Float16)v;
    }
    *(uint4*)&outF16[(size_t)(m0 + r) * 128 + ch * 8] = o.q;
  }
}

// ---------------------------------------------------------------------------
// Attention / softmax / scatter kernels
// ---------------------------------------------------------------------------
__global__ __launch_bounds__(256) void init_agg_kernel(float* agg, int n) {
  int i = blockIdx.x * 256 + threadIdx.x;
  if (i < n) agg[i] = 0.f;
}

__global__ __launch_bounds__(256) void init_seg_kernel(int* segMaxI, float* segSum, int n) {
  int i = blockIdx.x * 256 + threadIdx.x;
  if (i < n) { segMaxI[i] = (int)0x80000000; segSum[i] = 0.f; }
}

// score[e] = relu(edge[e] @ Wa + ba) @ Wh ; atomic max per dst segment
__global__ __launch_bounds__(256) void score_kernel(
    const _Float16* __restrict__ edge, const float* __restrict__ Wa,
    const float* __restrict__ ba, const float* __restrict__ Wh,
    const int* __restrict__ dstIdx, float* __restrict__ score,
    int* __restrict__ segMaxI, int E) {
  __shared__ float sWa[512];
  __shared__ float sBa[4], sWh[4];
  int tid = threadIdx.x;
  for (int i = tid; i < 512; i += 256) sWa[i] = Wa[i];
  if (tid < 4) { sBa[tid] = ba[tid]; sWh[tid] = Wh[tid]; }
  __syncthreads();
  int e = blockIdx.x * 256 + tid;
  if (e >= E) return;
  const _Float16* er = edge + (size_t)e * 128;
  float a0 = 0.f, a1 = 0.f, a2 = 0.f, a3 = 0.f;
  for (int k = 0; k < 128; ++k) {
    float xv = (float)er[k];
    a0 += xv * sWa[k * 4 + 0];
    a1 += xv * sWa[k * 4 + 1];
    a2 += xv * sWa[k * 4 + 2];
    a3 += xv * sWa[k * 4 + 3];
  }
  float s = fmaxf(a0 + sBa[0], 0.f) * sWh[0] + fmaxf(a1 + sBa[1], 0.f) * sWh[1] +
            fmaxf(a2 + sBa[2], 0.f) * sWh[2] + fmaxf(a3 + sBa[3], 0.f) * sWh[3];
  score[e] = s;
  atomicMax(&segMaxI[dstIdx[e]], f2ord(s));
}

__global__ __launch_bounds__(256) void expsum_kernel(
    const float* __restrict__ score, const int* __restrict__ segMaxI,
    const int* __restrict__ dstIdx, float* __restrict__ exv,
    float* __restrict__ segSum, int E) {
  int e = blockIdx.x * 256 + threadIdx.x;
  if (e >= E) return;
  int d = dstIdx[e];
  float m  = ord2f(segMaxI[d]);
  float ex = __expf(score[e] - m);
  exv[e] = ex;
  atomicAdd(&segSum[d], ex);
}

// agg[dst] += alpha[e] * new_edge[e]
__global__ __launch_bounds__(256) void scatter_kernel(
    const float* __restrict__ exv, const float* __restrict__ segSum,
    const int* __restrict__ dstIdx, const _Float16* __restrict__ newEdge,
    float* __restrict__ agg, int total) {
  int id = blockIdx.x * 256 + threadIdx.x;
  if (id >= total) return;
  int e = id >> 7, k = id & 127;
  int d = dstIdx[e];
  float alpha = exv[e] / segSum[d];
  atomicAdd(&agg[(size_t)d * 128 + k],
            alpha * (float)newEdge[(size_t)e * 128 + k]);
}

// ---------------------------------------------------------------------------
// Host launcher
// ---------------------------------------------------------------------------
extern "C" void kernel_launch(void* const* d_in, const int* in_sizes, int n_in,
                              void* d_out, int out_size, void* d_ws, size_t ws_size,
                              hipStream_t stream) {
  (void)in_sizes; (void)n_in; (void)out_size; (void)ws_size;
  const int N = N_NODES_C, E = N_EDGES_C;

  const float* x     = (const float*)d_in[0];
  const float* eattr = (const float*)d_in[1];
  const int*   eidx  = (const int*)d_in[2];
  const int*   srcI  = eidx;        // edge_index[0] = source (_j)
  const int*   dstI  = eidx + E;    // edge_index[1] = target (_i)
  auto P = [&](int i) { return (const float*)d_in[i]; };

  // Param leaf layout (insertion-order flatten of setup_inputs()):
  // fnet = Win, bin, W1, b1, W2, b2, Wout, bout [, lnG, lnB]
  // 3..12 node_enc | 13..22 edge_enc | 23..30 dec | 31 Wa, 32 ba, 33 Wh
  // 34..43 proc0.edge | 44..53 proc0.node | 54..63 proc1.edge | 64..73 proc1.node
  struct FD { int base, Kin, Nout; };
  const FD fds[7] = {{3, 16, 128}, {13, 8, 128}, {23, 128, 3},
                     {34, 384, 128}, {44, 256, 128},
                     {54, 384, 128}, {64, 256, 128}};

  // ---- workspace carve (deterministic) ----
  size_t off = 0;
  auto carve = [&](size_t bytes) {
    void* p = (char*)d_ws + off;
    off = (off + bytes + 255) & ~(size_t)255;
    return p;
  };

  unsigned* pW[7][4];
  for (int i = 0; i < 7; ++i) {
    int Kt = (fds[i].Kin + 31) / 32;
    int NT = (fds[i].Nout + 15) / 16;
    pW[i][0] = (unsigned*)carve((size_t)Kt * 8 * 256 * 4);   // Win
    pW[i][1] = (unsigned*)carve((size_t)4 * 8 * 256 * 4);    // W1
    pW[i][2] = (unsigned*)carve((size_t)4 * 8 * 256 * 4);    // W2
    pW[i][3] = (unsigned*)carve((size_t)4 * NT * 256 * 4);   // Wout
  }
  _Float16* nodeA = (_Float16*)carve((size_t)N * 128 * 2);
  _Float16* nodeB = (_Float16*)carve((size_t)N * 128 * 2);
  _Float16* edgeA = (_Float16*)carve((size_t)E * 128 * 2);
  _Float16* edgeB = (_Float16*)carve((size_t)E * 128 * 2);
  float* agg     = (float*)carve((size_t)N * 128 * 4);
  float* score   = (float*)carve((size_t)E * 4);
  float* exv     = (float*)carve((size_t)E * 4);
  int*   segMaxI = (int*)carve((size_t)N * 4);
  float* segSum  = (float*)carve((size_t)N * 4);

  // ---- pack all weights to WMMA f16 B-fragment order ----
  auto packOne = [&](const float* W, unsigned* dst, int K, int Nout) {
    int Kt = (K + 31) / 32, NT = (Nout + 15) / 16;
    int total = Kt * NT * 256;
    pack_weights_kernel<<<(total + 255) / 256, 256, 0, stream>>>(W, dst, K, Nout, NT, total);
  };
  for (int i = 0; i < 7; ++i) {
    int b = fds[i].base;
    packOne(P(b + 0), pW[i][0], fds[i].Kin, 128);
    packOne(P(b + 2), pW[i][1], 128, 128);
    packOne(P(b + 4), pW[i][2], 128, 128);
    packOne(P(b + 6), pW[i][3], 128, fds[i].Nout);
  }

  // ---- generic fnet launch ----
  auto fnet = [&](int mode, int M, int fi,
                  const float* rawIn, const _Float16* nodeF, const _Float16* edgeF,
                  const float* aggP, const _Float16* resP,
                  _Float16* o16, float* o32) {
    int b = fds[fi].base;
    int KtIn = (fds[fi].Kin + 31) / 32;
    int Nout = fds[fi].Nout;
    int NT = (Nout + 15) / 16;
    bool ln = (fi != 2);
    fnet_kernel<<<M / 16, 256, 0, stream>>>(
        mode, fds[fi].Kin, KtIn, rawIn, nodeF, edgeF, srcI, dstI, aggP,
        pW[fi][0], P(b + 1), pW[fi][1], P(b + 3), pW[fi][2], P(b + 5),
        pW[fi][3], P(b + 7), NT, Nout,
        ln ? P(b + 8) : nullptr, ln ? P(b + 9) : nullptr,
        resP, o16, o32);
  };

  // ---- encoders ----
  fnet(MODE_RAW, N, 0, x, nullptr, nullptr, nullptr, nullptr, nodeA, nullptr);
  fnet(MODE_RAW, E, 1, eattr, nullptr, nullptr, nullptr, nullptr, edgeA, nullptr);

  // ---- message passes ----
  const float* Wa = P(31);
  const float* ba = P(32);
  const float* Wh = P(33);
  for (int p = 0; p < 2; ++p) {
    const _Float16* nI = p ? nodeB : nodeA;
    _Float16*       nO = p ? nodeA : nodeB;
    const _Float16* eI = p ? edgeB : edgeA;
    _Float16*       eO = p ? edgeA : edgeB;
    int fe = 3 + 2 * p, fn = 4 + 2 * p;

    // new_edge = edge + edge_fnet([node[dst], node[src], edge])
    fnet(MODE_CEDGE, E, fe, nullptr, nI, eI, nullptr, eI, eO, nullptr);

    init_agg_kernel<<<(N * 128 + 255) / 256, 256, 0, stream>>>(agg, N * 128);
    init_seg_kernel<<<(N + 255) / 256, 256, 0, stream>>>(segMaxI, segSum, N);
    score_kernel<<<(E + 255) / 256, 256, 0, stream>>>(eI, Wa, ba, Wh, dstI, score, segMaxI, E);
    expsum_kernel<<<(E + 255) / 256, 256, 0, stream>>>(score, segMaxI, dstI, exv, segSum, E);
    scatter_kernel<<<(E * 128 + 255) / 256, 256, 0, stream>>>(exv, segSum, dstI, eO, agg, E * 128);

    // node = node + node_fnet([node, agg])
    fnet(MODE_CNODE, N, fn, nullptr, nI, nullptr, agg, nI, nO, nullptr);
  }

  // ---- decoder: out[N,3] f32 (final node latent is in nodeA) ----
  fnet(MODE_LATENT, N, 2, nullptr, nodeA, nullptr, nullptr, nullptr, nullptr, (float*)d_out);
}
